// MultiTaskMazeQNetwork_10136122818688
// MI455X (gfx1250) — matrix operand order repaired and checked by reading, not compiled, not run on hardware
//
// MI455X / gfx1250 multi-task Q-network forward.
// Pipeline (all on `stream`):
//   init_counts -> bucket_rows -> pack_bfrag x4 -> conv_fused (conv1 VALU +
//   conv2 WMMA) -> expbot (fused exp+bot WMMA) -> trep_gemm (task-bucketed
//   WMMA, 32 rows/block) -> head_k (VALU epilogue).
// All big GEMMs use v_wmma_f32_16x16x32_bf16 (wave32, 16x16 C tile, K=32).
// B-operands are pre-swizzled to the per-lane fragment layout so the GEMM
// inner loop is pure b128 loads + WMMA; K-loops are software-pipelined
// (double-buffered fragments) so WMMA overlaps the next step's loads.

#include <hip/hip_runtime.h>
#include <stdint.h>

typedef __attribute__((ext_vector_type(16))) __bf16 v16bf;
typedef __attribute__((ext_vector_type(8)))  float  v8f;

// Fragment viewed as two 16B quads for b128 loads.
union Frag {
  v16bf v;
  uint4 q[2];
};

__device__ __forceinline__ unsigned short f2bf(float f) {
  union { float f; uint32_t u; } x; x.f = f;
  uint32_t u = x.u;
  uint32_t r = u + 0x7FFFu + ((u >> 16) & 1u);   // round-to-nearest-even
  return (unsigned short)(r >> 16);
}

__device__ __forceinline__ v8f wmma_bf16(const Frag& a, const Frag& b, v8f c) {
  return __builtin_amdgcn_wmma_f32_16x16x32_bf16(
      /*neg_a=*/false, a.v, /*neg_b=*/false, b.v,
      /*c_mod=*/(short)0, c, /*reuse_a=*/false, /*reuse_b=*/false);
}

// A-fragment: lane holds K = klo..klo+7 (q0) and klo+16..klo+23 (q1)
__device__ __forceinline__ void ld_a(Frag& f, const unsigned short* p) {
  f.q[0] = *(const uint4*)(p);
  f.q[1] = *(const uint4*)(p + 16);
}
// Pre-swizzled B-fragment: 32 contiguous bytes per lane
__device__ __forceinline__ void ld_b(Frag& f, const unsigned short* p) {
  f.q[0] = *(const uint4*)(p);
  f.q[1] = *(const uint4*)(p + 8);
}

// ---------------------------------------------------------------------------
// Task bucketing: counts[t], rowidx[t*8192 + slot] = b
// ---------------------------------------------------------------------------
__global__ void init_counts(int* counts) {
  if (threadIdx.x < 16) counts[threadIdx.x] = 0;
}

__global__ __launch_bounds__(256) void bucket_rows(const int* __restrict__ task,
                                                   int* __restrict__ counts,
                                                   int* __restrict__ rowidx) {
  int b = blockIdx.x * 256 + threadIdx.x;
  if (b < 8192) {
    int t = task[b];
    int slot = atomicAdd(&counts[t], 1);
    rowidx[t * 8192 + slot] = b;
  }
}

// ---------------------------------------------------------------------------
// Pack an fp32 [K x Ncols] weight into bf16 WMMA B-fragments.
//   idx = ((kt*NT + nt)*32 + lane)*16 + e
//   Klocal = 8*(lane>>4) + (e&7) + 16*(e>>3) ; N = nt*16 + (lane&15)
// ---------------------------------------------------------------------------
__global__ __launch_bounds__(256) void pack_bfrag(const float* __restrict__ src,
                                                  unsigned short* __restrict__ dst,
                                                  int NT, int Ncols, int total) {
  int idx = blockIdx.x * 256 + threadIdx.x;
  if (idx >= total) return;
  int e    = idx & 15;
  int lane = (idx >> 4) & 31;
  int q    = idx >> 9;
  int nt   = q % NT;
  int kt   = q / NT;
  int K = kt * 32 + 8 * (lane >> 4) + (e & 7) + 16 * (e >> 3);
  int N = nt * 16 + (lane & 15);
  dst[idx] = f2bf(src[(size_t)K * Ncols + N]);
}

// ---------------------------------------------------------------------------
// conv1 (VALU fp32, zero-padded in LDS) + conv2 (WMMA implicit GEMM).
// One block = one image, 256 threads = 8 waves.
// conv2 as GEMM: C[64 spatial, 64 cout] = A[64, 512 patch] x W2[512, 64],
// K-tile of 32 == one (ky,kx) tap's 32 input channels (contiguous in LDS).
// Wave w owns M-tile (w>>1) and N-tiles (w&1)*2, (w&1)*2+1: one A frag
// feeds two WMMAs per k-step.
// ---------------------------------------------------------------------------
__global__ __launch_bounds__(256) void conv_fused(
    const float* __restrict__ x, const float* __restrict__ w1,
    const float* __restrict__ b1, const unsigned short* __restrict__ w2frag,
    const float* __restrict__ b2, unsigned short* __restrict__ flatbf) {
  __shared__ __align__(16) float xpad[18 * 18 * 3];            // x padded by 1
  __shared__ __align__(16) float w1s[48 * 32];                 // conv1 weights
  __shared__ __align__(16) unsigned short c1pad[18 * 18 * 32]; // conv1 out, pad 2

  const int tid = threadIdx.x;
  const int img = blockIdx.x;

  for (int i = tid; i < 18 * 18 * 3; i += 256) xpad[i] = 0.f;
  uint32_t* c1w = (uint32_t*)c1pad;
  for (int i = tid; i < (18 * 18 * 32) / 2; i += 256) c1w[i] = 0u;
  for (int i = tid; i < 48 * 32; i += 256) w1s[i] = w1[i];
  __syncthreads();

  for (int i = tid; i < 16 * 16 * 3; i += 256) {
    int ci = i % 3;
    int c  = (i / 3) % 16;
    int r  = i / 48;
    xpad[((r + 1) * 18 + (c + 1)) * 3 + ci] = x[(size_t)img * 768 + i];
  }
  __syncthreads();

  // conv1: 15x15x32 outputs, 48 MACs each, relu, store bf16 at pad-2 offset.
  for (int o = tid; o < 15 * 15 * 32; o += 256) {
    int co = o & 31;
    int j  = (o >> 5) % 15;
    int i  = o / (32 * 15);
    float acc = b1[co];
    for (int ky = 0; ky < 4; ++ky)
      for (int kx = 0; kx < 4; ++kx)
        for (int ci = 0; ci < 3; ++ci)
          acc += xpad[((i + ky) * 18 + (j + kx)) * 3 + ci] *
                 w1s[((ky * 4 + kx) * 3 + ci) * 32 + co];
    c1pad[((i + 2) * 18 + (j + 2)) * 32 + co] = f2bf(fmaxf(acc, 0.f));
  }
  __syncthreads();

  // conv2 WMMA, software-pipelined over the 16 k-steps.
  const int lane = tid & 31, wave = tid >> 5;
  const int klo = 8 * (lane >> 4);
  const int mt  = wave >> 1;
  const int nt0 = (wave & 1) * 2;
  const int m   = lane & 15;
  const int s   = mt * 16 + m;
  const int si = s >> 3, sj = s & 7;

  auto a_addr = [&](int kk) -> const unsigned short* {
    int ky = kk >> 2, kx = kk & 3;
    return &c1pad[((2 * si + ky) * 18 + (2 * sj + kx)) * 32 + klo];
  };
  auto b_addr = [&](int kk, int n) -> const unsigned short* {
    return &w2frag[(((kk * 4 + nt0 + n) * 32) + lane) * 16];
  };

  v8f acc[2] = {};
  Frag ac, bc0, bc1;
  ld_a(ac, a_addr(0));
  ld_b(bc0, b_addr(0, 0));
  ld_b(bc1, b_addr(0, 1));
#pragma unroll 4
  for (int kk = 0; kk < 16; ++kk) {
    Frag an, bn0, bn1;
    const int kn = (kk + 1 < 16) ? kk + 1 : kk;
    ld_a(an, a_addr(kn));
    ld_b(bn0, b_addr(kn, 0));
    ld_b(bn1, b_addr(kn, 1));
    acc[0] = wmma_bf16(ac, bc0, acc[0]);
    acc[1] = wmma_bf16(ac, bc1, acc[1]);
    ac = an; bc0 = bn0; bc1 = bn1;
  }

  // epilogue: bias + relu -> flat bf16 (layout (i*8+j)*64 + cout)
  const int mhalf = 8 * (lane >> 4);
  for (int n = 0; n < 2; ++n) {
    const int nlo = (nt0 + n) * 16 + (lane & 15);
    const float bias = b2[nlo];
    for (int r = 0; r < 8; ++r) {
      int ss = mt * 16 + mhalf + r;
      flatbf[(size_t)img * 4096 + ss * 64 + nlo] =
          f2bf(fmaxf(acc[n][r] + bias, 0.f));
    }
  }
}

// ---------------------------------------------------------------------------
// Fused exp (4096->512, +bias, NO relu) then bot (512->128, +bias, relu).
// One block = 16 rows. Phase 1: wave w computes cols 64w..64w+63 of hidden
// (4 n-tiles, pipelined K loop of 128). Hidden staged in LDS bf16.
// Phase 2: wave w computes cols 16w..16w+15 of shared (pipelined K loop 16).
// ---------------------------------------------------------------------------
__global__ __launch_bounds__(256) void expbot(
    const unsigned short* __restrict__ flatbf,
    const unsigned short* __restrict__ expfrag, const float* __restrict__ exp_b,
    const unsigned short* __restrict__ botfrag, const float* __restrict__ bot_b,
    float* __restrict__ shared_out) {
  __shared__ __align__(16) unsigned short hid[16 * 512];

  const int tid = threadIdx.x, lane = tid & 31, wave = tid >> 5;
  const int row0 = blockIdx.x * 16;
  const int klo = 8 * (lane >> 4);
  const int mhalf = 8 * (lane >> 4);

  // ---- phase 1: exp ----
  const unsigned short* abase = flatbf + (size_t)(row0 + (lane & 15)) * 4096;
  auto a_addr = [&](int kt) { return abase + kt * 32 + klo; };
  auto b_addr = [&](int kt, int n) {
    return expfrag + ((size_t)(kt * 32 + wave * 4 + n) * 32 + lane) * 16;
  };

  v8f acc[4] = {};
  Frag ac, bc[4];
  ld_a(ac, a_addr(0));
  for (int n = 0; n < 4; ++n) ld_b(bc[n], b_addr(0, n));
#pragma unroll 2
  for (int kt = 0; kt < 128; ++kt) {
    Frag an, bn[4];
    const int kn = (kt + 1 < 128) ? kt + 1 : kt;
    ld_a(an, a_addr(kn));
    for (int n = 0; n < 4; ++n) ld_b(bn[n], b_addr(kn, n));
    if (kt + 8 < 128) __builtin_prefetch(b_addr(kt + 8, 0), 0, 0);
    for (int n = 0; n < 4; ++n) acc[n] = wmma_bf16(ac, bc[n], acc[n]);
    ac = an;
    for (int n = 0; n < 4; ++n) bc[n] = bn[n];
  }
  for (int n = 0; n < 4; ++n) {
    const int col = wave * 64 + n * 16 + (lane & 15);
    const float bias = exp_b[col];
    for (int r = 0; r < 8; ++r)
      hid[(mhalf + r) * 512 + col] = f2bf(acc[n][r] + bias);  // no relu here
  }
  __syncthreads();

  // ---- phase 2: bot ----
  const int mm = lane & 15;
  auto a2_addr = [&](int kt) { return &hid[mm * 512 + kt * 32 + klo]; };
  auto b2_addr = [&](int kt) {
    return botfrag + ((size_t)(kt * 8 + wave) * 32 + lane) * 16;
  };
  v8f c = {};
  Frag a2c, b2c;
  ld_a(a2c, a2_addr(0));
  ld_b(b2c, b2_addr(0));
#pragma unroll 4
  for (int kt = 0; kt < 16; ++kt) {
    Frag a2n, b2n;
    const int kn = (kt + 1 < 16) ? kt + 1 : kt;
    ld_a(a2n, a2_addr(kn));
    ld_b(b2n, b2_addr(kn));
    c = wmma_bf16(a2c, b2c, c);
    a2c = a2n; b2c = b2n;
  }
  const int col = wave * 16 + (lane & 15);
  const float bias = bot_b[col];
  for (int r = 0; r < 8; ++r)
    shared_out[(size_t)(row0 + mhalf + r) * 128 + col] = fmaxf(c[r] + bias, 0.f);
}

// ---------------------------------------------------------------------------
// Task-bucketed trep GEMM: for task t, gathered rows x trep_w[t] (4096->128).
// Grid = 16 tasks x 256 row-blocks of 32 rows; blocks past the bucket count
// exit early. Wave w owns n-tile w and BOTH m-tiles: one B fragment feeds
// two WMMAs per k-step (halves the dominant weight stream).
// ---------------------------------------------------------------------------
__global__ __launch_bounds__(256) void trep_gemm(
    const unsigned short* __restrict__ flatbf,
    const unsigned short* __restrict__ trepfrag, const float* __restrict__ trep_b,
    const int* __restrict__ counts, const int* __restrict__ rowidx,
    float* __restrict__ trep_out) {
  const int t  = blockIdx.x >> 8;
  const int mb = blockIdx.x & 255;
  const int cnt = counts[t];
  if (mb * 32 >= cnt) return;

  __shared__ int rows[32];
  const int tid = threadIdx.x, lane = tid & 31, wave = tid >> 5;
  if (tid < 32) {
    int idx = mb * 32 + tid;
    rows[tid] = (idx < cnt) ? rowidx[t * 8192 + idx] : -1;
  }
  __syncthreads();

  int ar0 = rows[lane & 15];        if (ar0 < 0) ar0 = 0;
  int ar1 = rows[16 + (lane & 15)]; if (ar1 < 0) ar1 = 0;
  const unsigned short* ab0 = flatbf + (size_t)ar0 * 4096;
  const unsigned short* ab1 = flatbf + (size_t)ar1 * 4096;
  const unsigned short* bbase = trepfrag + (size_t)t * (128 * 8 * 32 * 16);
  const int klo = 8 * (lane >> 4);

  auto b_addr = [&](int kt) {
    return bbase + ((size_t)(kt * 8 + wave) * 32 + lane) * 16;
  };

  v8f c0 = {}, c1 = {};
  Frag a0c, a1c, bc;
  ld_a(a0c, ab0 + klo);
  ld_a(a1c, ab1 + klo);
  ld_b(bc, b_addr(0));
#pragma unroll 2
  for (int kt = 0; kt < 128; ++kt) {
    Frag a0n, a1n, bn;
    const int kn = (kt + 1 < 128) ? kt + 1 : kt;
    ld_a(a0n, ab0 + kn * 32 + klo);
    ld_a(a1n, ab1 + kn * 32 + klo);
    ld_b(bn, b_addr(kn));
    if (kt + 8 < 128) __builtin_prefetch(b_addr(kt + 8), 0, 0);
    c0 = wmma_bf16(a0c, bc, c0);
    c1 = wmma_bf16(a1c, bc, c1);
    a0c = a0n; a1c = a1n; bc = bn;
  }
  const int col = wave * 16 + (lane & 15);
  const float bias = trep_b[t * 128 + col];
  const int mhalf = 8 * (lane >> 4);
  for (int r = 0; r < 8; ++r) {
    int row0 = rows[mhalf + r];
    if (row0 >= 0)
      trep_out[(size_t)row0 * 128 + col] = fmaxf(c0[r] + bias, 0.f);
    int row1 = rows[16 + mhalf + r];
    if (row1 >= 0)
      trep_out[(size_t)row1 * 128 + col] = fmaxf(c1[r] + bias, 0.f);
  }
}

// ---------------------------------------------------------------------------
// Head: out[b,:] = thead_b[t] + [shared(128) | trep(128)] @ thead_w[t]
// Tiny (16 MFLOP) -> plain VALU, one thread per row.
// ---------------------------------------------------------------------------
__global__ __launch_bounds__(256) void head_k(
    const float* __restrict__ shared_out, const float* __restrict__ trep_out,
    const int* __restrict__ task, const float* __restrict__ thw,
    const float* __restrict__ thb, float* __restrict__ out) {
  int b = blockIdx.x * 256 + threadIdx.x;
  if (b >= 8192) return;
  const int t = task[b];
  const float* w = thw + (size_t)t * 256 * 4;
  float a0 = thb[t * 4 + 0], a1 = thb[t * 4 + 1];
  float a2 = thb[t * 4 + 2], a3 = thb[t * 4 + 3];
  for (int c = 0; c < 128; ++c) {
    float s = shared_out[(size_t)b * 128 + c];
    a0 += s * w[c * 4 + 0]; a1 += s * w[c * 4 + 1];
    a2 += s * w[c * 4 + 2]; a3 += s * w[c * 4 + 3];
  }
  for (int c = 0; c < 128; ++c) {
    float s = trep_out[(size_t)b * 128 + c];
    a0 += s * w[(128 + c) * 4 + 0]; a1 += s * w[(128 + c) * 4 + 1];
    a2 += s * w[(128 + c) * 4 + 2]; a3 += s * w[(128 + c) * 4 + 3];
  }
  out[b * 4 + 0] = a0; out[b * 4 + 1] = a1;
  out[b * 4 + 2] = a2; out[b * 4 + 3] = a3;
}

// ---------------------------------------------------------------------------
extern "C" void kernel_launch(void* const* d_in, const int* in_sizes, int n_in,
                              void* d_out, int out_size, void* d_ws, size_t ws_size,
                              hipStream_t stream) {
  const float* x    = (const float*)d_in[0];
  const int*   task = (const int*)d_in[1];
  const float* w1   = (const float*)d_in[2];
  const float* b1   = (const float*)d_in[3];
  const float* w2   = (const float*)d_in[4];
  const float* b2   = (const float*)d_in[5];
  const float* expw = (const float*)d_in[6];
  const float* expb = (const float*)d_in[7];
  const float* botw = (const float*)d_in[8];
  const float* botb = (const float*)d_in[9];
  const float* trw  = (const float*)d_in[10];
  const float* trb  = (const float*)d_in[11];
  const float* thw  = (const float*)d_in[12];
  const float* thb  = (const float*)d_in[13];
  float* out = (float*)d_out;

  char* ws = (char*)d_ws;
  size_t off = 0;
  auto take = [&](size_t bytes) -> void* {
    void* p = ws + off;
    off = (off + bytes + 255) & ~(size_t)255;
    return p;
  };
  unsigned short* flatbf   = (unsigned short*)take((size_t)8192 * 4096 * 2); // 64 MB
  float* shared_out        = (float*)take((size_t)8192 * 128 * 4);           // 4 MB
  float* trep_out          = (float*)take((size_t)8192 * 128 * 4);           // 4 MB
  int* counts              = (int*)take(64);
  int* rowidx              = (int*)take((size_t)16 * 8192 * 4);              // 512 KB
  unsigned short* expfrag  = (unsigned short*)take((size_t)4096 * 512 * 2);  // 4 MB
  unsigned short* botfrag  = (unsigned short*)take((size_t)512 * 128 * 2);   // 128 KB
  unsigned short* trepfrag = (unsigned short*)take((size_t)16 * 4096 * 128 * 2); // 16 MB
  unsigned short* w2frag   = (unsigned short*)take((size_t)32768 * 2);       // 64 KB

  init_counts<<<1, 32, 0, stream>>>(counts);
  bucket_rows<<<32, 256, 0, stream>>>(task, counts, rowidx);
  pack_bfrag<<<8192, 256, 0, stream>>>(expw, expfrag, 32, 512, 2097152);
  pack_bfrag<<<256, 256, 0, stream>>>(botw, botfrag, 8, 128, 65536);
  pack_bfrag<<<32768, 256, 0, stream>>>(trw, trepfrag, 8, 128, 8388608);
  pack_bfrag<<<128, 256, 0, stream>>>(w2, w2frag, 4, 64, 32768);
  conv_fused<<<8192, 256, 0, stream>>>(x, w1, b1, w2frag, b2, flatbf);
  expbot<<<512, 256, 0, stream>>>(flatbf, expfrag, expb, botfrag, botb, shared_out);
  trep_gemm<<<4096, 256, 0, stream>>>(flatbf, trepfrag, trb, counts, rowidx, trep_out);
  head_k<<<32, 256, 0, stream>>>(shared_out, trep_out, task, thw, thb, out);
}
